// GATLayer_1_39977555591204
// MI455X (gfx1250) — matrix operand order repaired and checked by reading, not compiled, not run on hardware
//
#include <hip/hip_runtime.h>
#include <cstdint>
#include <cstddef>

// ---------------------------------------------------------------------------
// GAT (2-layer, PyG GATConv semantics) for MI455X / gfx1250.
// Projections use v_wmma_f32_16x16x32_f16; graph phases use HW float atomics
// (inline-asm global_atomic_add_f32 at device scope to guarantee no CAS loop).
// ---------------------------------------------------------------------------

#define NEG_SLOPE 0.2f

typedef __attribute__((ext_vector_type(16))) _Float16 v16h;
typedef __attribute__((ext_vector_type(8)))  _Float16 v8h;
typedef __attribute__((ext_vector_type(8)))  float    v8f;

// ---- native fp32 atomic add (ISA §15.18.3 op 86), no-return form ----------
__device__ __forceinline__ void atomicAddF(float* p, float v) {
#if defined(__gfx1250__)
  asm volatile("global_atomic_add_f32 %0, %1, off scope:SCOPE_DEV"
               :: "v"(p), "v"(v) : "memory");
#else
  atomicAdd(p, v);
#endif
}

// ---- order-preserving float <-> uint encoding for atomic max -------------
__device__ __forceinline__ unsigned fenc(float f) {
  unsigned u = __float_as_uint(f);
  return (u & 0x80000000u) ? ~u : (u | 0x80000000u);
}
__device__ __forceinline__ float fdec(unsigned k) {
  unsigned u = (k & 0x80000000u) ? (k & 0x7fffffffu) : ~k;
  return __uint_as_float(u);
}

// ---- init ----------------------------------------------------------------
__global__ void fill_u32_kernel(unsigned* __restrict__ p, unsigned v, size_t n) {
  size_t i = (size_t)blockIdx.x * blockDim.x + threadIdx.x;
  if (i < n) p[i] = v;
}

// ---- degree + incoming edge_attr sum -------------------------------------
__global__ void deg_kernel(const int* __restrict__ dst, const float* __restrict__ ew,
                           float* __restrict__ deg, float* __restrict__ lsum, int E) {
  int e = blockIdx.x * blockDim.x + threadIdx.x;
  if (e >= E) return;
  int d = dst[e];
  atomicAddF(&deg[d], 1.0f);
  atomicAddF(&lsum[d], ew[e]);
}

__global__ void loopattr_kernel(const float* __restrict__ deg, const float* __restrict__ lsum,
                                float* __restrict__ la, int N) {
  int i = blockIdx.x * blockDim.x + threadIdx.x;
  if (i >= N) return;
  la[i] = lsum[i] / fmaxf(deg[i], 1.0f);
}

// ---- edge-attention scalar c[h] = sum_c We[h*C+c]*ae[h*C+c] --------------
__global__ void cvec_kernel(const float* __restrict__ We, const float* __restrict__ ae,
                            float* __restrict__ out, int H, int C) {
  int h = threadIdx.x;
  if (h >= H) return;
  float s = 0.f;
  for (int c = 0; c < C; ++c) s += We[h * C + c] * ae[h * C + c];
  out[h] = s;
}

// ---- fp32 -> fp16 conversions --------------------------------------------
__global__ void cvt_f16_kernel(const float* __restrict__ in, _Float16* __restrict__ out, size_t n) {
  size_t i = (size_t)blockIdx.x * blockDim.x + threadIdx.x;
  if (i < n) out[i] = (_Float16)in[i];
}

// transpose-convert: W[K][Nc] (row major) -> Wt[Nc][K] fp16
__global__ void cvt_wT_kernel(const float* __restrict__ W, _Float16* __restrict__ Wt,
                              int K, int Nc) {
  int idx = blockIdx.x * blockDim.x + threadIdx.x;
  if (idx >= K * Nc) return;
  int k = idx / Nc, n = idx % Nc;
  Wt[(size_t)n * K + k] = (_Float16)W[idx];
}

// ---- WMMA GEMM: C[M, Nc] = A[M, K](f16) x Bt[Nc, K](f16), f32 accumulate --
// One wave computes one 16x16 tile. A-fragment per ISA 16-bit A 16x32 layout,
// B-fragment per 16-bit B layout (lane = column, same K order as A).
template <int K>
__global__ __launch_bounds__(256) void gemm_wmma_kernel(
    const _Float16* __restrict__ A, const _Float16* __restrict__ Bt,
    float* __restrict__ C, int Mtiles, int Ntiles) {
  int wave = blockIdx.x * (blockDim.x >> 5) + (threadIdx.x >> 5);
  if (wave >= Mtiles * Ntiles) return;     // wave-uniform: EXEC stays all-ones
  int lane = threadIdx.x & 31;
  int tm = wave / Ntiles, tn = wave % Ntiles;
  int r  = lane & 15;        // row (A) / column (B) within tile
  int hw = lane >> 4;        // half-wave select

  const _Float16* ap = A  + (size_t)(tm * 16 + r) * K + hw * 8;
  const _Float16* bp = Bt + (size_t)(tn * 16 + r) * K + hw * 8;

  v8f acc = {};
#pragma unroll
  for (int k = 0; k < K; k += 32) {
    v8h alo = *(const v8h*)(ap + k);        // K = k+hw*8 .. +7
    v8h ahi = *(const v8h*)(ap + k + 16);   // K = k+16+hw*8 .. +7
    v16h a = __builtin_shufflevector(alo, ahi, 0, 1, 2, 3, 4, 5, 6, 7,
                                               8, 9, 10, 11, 12, 13, 14, 15);
    v8h blo = *(const v8h*)(bp + k);        // K = k+hw*8 .. +7
    v8h bhi = *(const v8h*)(bp + k + 16);   // K = k+16+hw*8 .. +7
    v16h b = __builtin_shufflevector(blo, bhi, 0, 1, 2, 3, 4, 5, 6, 7,
                                               8, 9, 10, 11, 12, 13, 14, 15);
    acc = __builtin_amdgcn_wmma_f32_16x16x32_f16(false, a, false, b,
                                                 (short)0, acc, false, false);
  }
  // D layout: VGPR i -> M = i + 8*hw ; N = lane&15
  int Nc = Ntiles * 16;
  float* cp = C + (size_t)(tm * 16 + hw * 8) * Nc + tn * 16 + r;
#pragma unroll
  for (int i = 0; i < 8; ++i) cp[(size_t)i * Nc] = acc[i];
}

// ---- per-node attention dots ---------------------------------------------
template <int H, int C>
__global__ void node_attn_kernel(const float* __restrict__ feat,
                                 const float* __restrict__ a_src,
                                 const float* __restrict__ a_dst,
                                 float* __restrict__ alS, float* __restrict__ alD, int N) {
  int i = blockIdx.x * blockDim.x + threadIdx.x;
  if (i >= N) return;
  const float* fr = feat + (size_t)i * H * C;
#pragma unroll
  for (int h = 0; h < H; ++h) {
    float s = 0.f, d = 0.f;
    for (int c = 0; c < C; ++c) {
      float v = fr[h * C + c];
      s += v * a_src[h * C + c];
      d += v * a_dst[h * C + c];
    }
    alS[i * H + h] = s;
    alD[i * H + h] = d;
  }
}

// ---- edge pass 1: leaky-relu logit + segment max (encoded uint atomics) ---
template <int H>
__global__ void edge_max_kernel(const int* __restrict__ src, const int* __restrict__ dst,
                                const float* __restrict__ ew, const float* __restrict__ la,
                                const float* __restrict__ alS, const float* __restrict__ alD,
                                const float* __restrict__ cvec, unsigned* __restrict__ mkeys,
                                int E, int N) {
  int e = blockIdx.x * blockDim.x + threadIdx.x;
  if (e >= E + N) return;
  int s, d; float ea;
  if (e < E) { s = src[e]; d = dst[e]; ea = ew[e]; }
  else       { s = d = e - E; ea = la[s]; }
#pragma unroll
  for (int h = 0; h < H; ++h) {
    float l = alS[s * H + h] + alD[d * H + h] + ea * cvec[h];
    l = (l > 0.f) ? l : NEG_SLOPE * l;
    atomicMax(&mkeys[d * H + h], fenc(l));
  }
}

// ---- edge pass 2: p = exp(l - m), segment sum -----------------------------
template <int H>
__global__ void edge_p_kernel(const int* __restrict__ src, const int* __restrict__ dst,
                              const float* __restrict__ ew, const float* __restrict__ la,
                              const float* __restrict__ alS, const float* __restrict__ alD,
                              const float* __restrict__ cvec, const unsigned* __restrict__ mkeys,
                              float* __restrict__ p, float* __restrict__ denom, int E, int N) {
  int e = blockIdx.x * blockDim.x + threadIdx.x;
  if (e >= E + N) return;
  int s, d; float ea;
  if (e < E) { s = src[e]; d = dst[e]; ea = ew[e]; }
  else       { s = d = e - E; ea = la[s]; }
#pragma unroll
  for (int h = 0; h < H; ++h) {
    float l = alS[s * H + h] + alD[d * H + h] + ea * cvec[h];
    l = (l > 0.f) ? l : NEG_SLOPE * l;
    float pe = expf(l - fdec(mkeys[d * H + h]));
    p[(size_t)e * H + h] = pe;
    atomicAddF(&denom[d * H + h], pe);
  }
}

// ---- edge pass 3: out[dst] += alpha * feat[src]; one wave per edge --------
template <int H, int CT>
__global__ __launch_bounds__(256) void edge_agg_kernel(
    const int* __restrict__ src, const int* __restrict__ dst,
    const float* __restrict__ p, const float* __restrict__ denom,
    const float* __restrict__ feat, float* __restrict__ out, int E, int N) {
  int wave = blockIdx.x * (blockDim.x >> 5) + (threadIdx.x >> 5);
  if (wave >= E + N) return;
  int lane = threadIdx.x & 31;
  int e = wave, s, d;
  if (e < E) { s = src[e]; d = dst[e]; } else { s = d = e - E; }
  float alpha[H];
#pragma unroll
  for (int h = 0; h < H; ++h)
    alpha[h] = p[(size_t)e * H + h] / (denom[d * H + h] + 1e-16f);
  const float* fr = feat + (size_t)s * CT;
  float* orow = out + (size_t)d * CT;
#pragma unroll
  for (int f = lane; f < CT; f += 32)
    atomicAddF(&orow[f], alpha[f / (CT / H)] * fr[f]);
}

// ---- layer-1 epilogue: +bias, ELU in place, also emit f16 copy ------------
__global__ void finalize1_kernel(float* __restrict__ out1, const float* __restrict__ b1,
                                 _Float16* __restrict__ h16, size_t n, int C) {
  size_t i = (size_t)blockIdx.x * blockDim.x + threadIdx.x;
  if (i >= n) return;
  float v = out1[i] + b1[i % C];
  v = (v > 0.f) ? v : (expf(v) - 1.0f);
  out1[i] = v;
  h16[i]  = (_Float16)v;
}

// ---- layer-2 epilogue: +bias -> d_out ------------------------------------
__global__ void finalize2_kernel(const float* __restrict__ out2, const float* __restrict__ b2,
                                 float* __restrict__ y, size_t n, int C) {
  size_t i = (size_t)blockIdx.x * blockDim.x + threadIdx.x;
  if (i >= n) return;
  y[i] = out2[i] + b2[i % C];
}

// ===========================================================================
extern "C" void kernel_launch(void* const* d_in, const int* in_sizes, int n_in,
                              void* d_out, int out_size, void* d_ws, size_t ws_size,
                              hipStream_t stream) {
  const float* x   = (const float*)d_in[0];
  const int*   ei  = (const int*)d_in[1];
  const float* ew  = (const float*)d_in[2];
  const float* W1  = (const float*)d_in[3];
  const float* as1 = (const float*)d_in[4];
  const float* ad1 = (const float*)d_in[5];
  const float* We1 = (const float*)d_in[6];
  const float* ae1 = (const float*)d_in[7];
  const float* b1  = (const float*)d_in[8];
  const float* W2  = (const float*)d_in[9];
  const float* as2 = (const float*)d_in[10];
  const float* ad2 = (const float*)d_in[11];
  const float* We2 = (const float*)d_in[12];
  const float* ae2 = (const float*)d_in[13];
  const float* b2  = (const float*)d_in[14];

  const int IN = 128, HID = 64, HEADS = 2, OUT = 32;
  const int C1 = HEADS * HID;                 // 128
  const int N = in_sizes[0] / IN;             // 50000
  const int E = in_sizes[1] / 2;              // 1600000
  const int EN = E + N;
  const int* srcp = ei;
  const int* dstp = ei + E;

  // ---- workspace layout --------------------------------------------------
  char* ws = (char*)d_ws;
  size_t off = 0;
  auto alloc = [&](size_t bytes) -> char* {
    off = (off + 255) & ~(size_t)255;
    char* p = ws + off;
    off += bytes;
    return p;
  };
  float*    deg    = (float*)alloc((size_t)N * 4);
  float*    lsum   = (float*)alloc((size_t)N * 4);
  float*    la     = (float*)alloc((size_t)N * 4);
  float*    cvec   = (float*)alloc(16);                       // [0..1]=L1, [2]=L2
  float*    alS1   = (float*)alloc((size_t)N * HEADS * 4);
  float*    alD1   = (float*)alloc((size_t)N * HEADS * 4);
  unsigned* m1     = (unsigned*)alloc((size_t)N * HEADS * 4);
  float*    den1   = (float*)alloc((size_t)N * HEADS * 4);
  _Float16* W1t    = (_Float16*)alloc((size_t)IN * C1 * 2);
  _Float16* W2t    = (_Float16*)alloc((size_t)C1 * OUT * 2);
  _Float16* xh     = (_Float16*)alloc((size_t)N * IN * 2);    // reused as hact_f16
  float*    h1     = (float*)alloc((size_t)N * C1 * 4);
  float*    out1   = (float*)alloc((size_t)N * C1 * 4);       // reused as hact_f32
  float*    p1     = (float*)alloc((size_t)EN * HEADS * 4);
  float*    h2     = (float*)alloc((size_t)N * OUT * 4);
  float*    out2   = (float*)alloc((size_t)N * OUT * 4);
  float*    p2     = (float*)alloc((size_t)EN * 4);
  float*    alS2   = (float*)alloc((size_t)N * 4);
  float*    alD2   = (float*)alloc((size_t)N * 4);
  unsigned* m2     = (unsigned*)alloc((size_t)N * 4);
  float*    den2   = (float*)alloc((size_t)N * 4);
  (void)ws_size; (void)n_in; (void)out_size;

  const int B = 256;
  auto gsz = [&](size_t n) { return (unsigned)((n + B - 1) / B); };

  // ---- init accumulators (encoded -inf == 0 for max keys) ----------------
  fill_u32_kernel<<<gsz(N), B, 0, stream>>>((unsigned*)deg, 0u, (size_t)N);
  fill_u32_kernel<<<gsz(N), B, 0, stream>>>((unsigned*)lsum, 0u, (size_t)N);
  fill_u32_kernel<<<gsz((size_t)N * HEADS), B, 0, stream>>>(m1, 0u, (size_t)N * HEADS);
  fill_u32_kernel<<<gsz((size_t)N * HEADS), B, 0, stream>>>((unsigned*)den1, 0u, (size_t)N * HEADS);
  fill_u32_kernel<<<gsz((size_t)N * C1), B, 0, stream>>>((unsigned*)out1, 0u, (size_t)N * C1);
  fill_u32_kernel<<<gsz(N), B, 0, stream>>>(m2, 0u, (size_t)N);
  fill_u32_kernel<<<gsz(N), B, 0, stream>>>((unsigned*)den2, 0u, (size_t)N);
  fill_u32_kernel<<<gsz((size_t)N * OUT), B, 0, stream>>>((unsigned*)out2, 0u, (size_t)N * OUT);

  // ---- self-loop attrs + edge-attn scalars -------------------------------
  deg_kernel<<<gsz(E), B, 0, stream>>>(dstp, ew, deg, lsum, E);
  loopattr_kernel<<<gsz(N), B, 0, stream>>>(deg, lsum, la, N);
  cvec_kernel<<<1, HEADS, 0, stream>>>(We1, ae1, cvec, HEADS, HID);
  cvec_kernel<<<1, 1, 0, stream>>>(We2, ae2, cvec + 2, 1, OUT);

  // ---- f16 conversions ---------------------------------------------------
  cvt_f16_kernel<<<gsz((size_t)N * IN), B, 0, stream>>>(x, xh, (size_t)N * IN);
  cvt_wT_kernel<<<gsz((size_t)IN * C1), B, 0, stream>>>(W1, W1t, IN, C1);
  cvt_wT_kernel<<<gsz((size_t)C1 * OUT), B, 0, stream>>>(W2, W2t, C1, OUT);

  // ---- layer 1 -----------------------------------------------------------
  {
    int Mt = N / 16, Nt = C1 / 16;                 // 3125 x 8 tiles
    int waves = Mt * Nt, blocks = (waves + 7) / 8;
    gemm_wmma_kernel<128><<<blocks, B, 0, stream>>>(xh, W1t, h1, Mt, Nt);
  }
  node_attn_kernel<2, 64><<<gsz(N), B, 0, stream>>>(h1, as1, ad1, alS1, alD1, N);
  edge_max_kernel<2><<<gsz(EN), B, 0, stream>>>(srcp, dstp, ew, la, alS1, alD1, cvec, m1, E, N);
  edge_p_kernel<2><<<gsz(EN), B, 0, stream>>>(srcp, dstp, ew, la, alS1, alD1, cvec, m1, p1, den1, E, N);
  edge_agg_kernel<2, 128><<<(EN + 7) / 8, B, 0, stream>>>(srcp, dstp, p1, den1, h1, out1, E, N);
  finalize1_kernel<<<gsz((size_t)N * C1), B, 0, stream>>>(out1, b1, xh, (size_t)N * C1, C1);

  // ---- layer 2 (hact f32 = out1 in place, hact f16 = xh in place) --------
  {
    int Mt = N / 16, Nt = OUT / 16;                // 3125 x 2 tiles
    int waves = Mt * Nt, blocks = (waves + 7) / 8;
    gemm_wmma_kernel<128><<<blocks, B, 0, stream>>>(xh, W2t, h2, Mt, Nt);
  }
  node_attn_kernel<1, 32><<<gsz(N), B, 0, stream>>>(h2, as2, ad2, alS2, alD2, N);
  edge_max_kernel<1><<<gsz(EN), B, 0, stream>>>(srcp, dstp, ew, la, alS2, alD2, cvec + 2, m2, E, N);
  edge_p_kernel<1><<<gsz(EN), B, 0, stream>>>(srcp, dstp, ew, la, alS2, alD2, cvec + 2, m2, p2, den2, E, N);
  edge_agg_kernel<1, 32><<<(EN + 7) / 8, B, 0, stream>>>(srcp, dstp, p2, den2, h2, out2, E, N);
  finalize2_kernel<<<gsz((size_t)N * OUT), B, 0, stream>>>(out2, b2, (float*)d_out, (size_t)N * OUT, OUT);
}
